// MaskRCNN_63917703299857
// MI455X (gfx1250) — matrix operand order repaired and checked by reading, not compiled, not run on hardware
//
#include <hip/hip_runtime.h>
#include <math.h>

// ---------------------------------------------------------------- types ----
typedef __attribute__((ext_vector_type(16))) __bf16         v16bf;
typedef __attribute__((ext_vector_type(16))) unsigned short v16us;
typedef __attribute__((ext_vector_type(8)))  unsigned short v8us;
typedef __attribute__((ext_vector_type(8)))  float          v8f;

union FragAB { v16us u; v16bf b; };

__device__ inline unsigned short f2bf(float f) {
  unsigned int u = __float_as_uint(f);
  if ((u & 0x7fffffffu) > 0x7f800000u) return (unsigned short)((u >> 16) | 0x0040u); // quiet NaN
  u += 0x7fffu + ((u >> 16) & 1u);   // round-to-nearest-even
  return (unsigned short)(u >> 16);
}

// A-fragment index for element (m,k): ISA 7.12.2 16-bit A 16x32 layout.
//   half = (k>>3)&1 (within 32-chunk), lane = 16*half + (m&15), e = (k&7)|((k&16)>>1)
__device__ inline size_t a_frag_idx(int m, int k, int Kt) {
  int kt = k >> 5, kcl = k & 31;
  int half = (kcl >> 3) & 1;
  int lane = 16 * half + (m & 15);
  int e = (kcl & 7) | ((kcl & 16) >> 1);
  return (((size_t)(m >> 4) * Kt + kt) * 32 + lane) * 16 + e;
}

// B-fragment index for element (k,n): same k order as A: e = (k&7)|((k&16)>>1), half = (k>>3)&1.
__device__ inline size_t b_frag_idx(int k, int n, int Nt) {
  int kt = k >> 5, kcl = k & 31;
  int lane = 16 * ((kcl >> 3) & 1) + (n & 15);
  int e = (kcl & 7) | ((kcl & 16) >> 1);
  return (((size_t)kt * Nt + (n >> 4)) * 32 + lane) * 16 + e;
}

// ---------------------------------------------------------------- packs ----
// f32 CHW -> bf16 NHWC (so 32 consecutive GEMM-K = 32 channels are contiguous)
__global__ void k_pack_nhwc(const float* __restrict__ src, unsigned short* __restrict__ dst,
                            int HW) {
  int i = blockIdx.x * blockDim.x + threadIdx.x;          // i = pos*256 + c
  if (i >= HW * 256) return;
  int c = i & 255, pos = i >> 8;
  dst[i] = f2bf(src[c * HW + pos]);
}

// rpn_conv_w (512,256,3,3) OIHW -> B-fragment layout, k = tap*256+c, Nld=512 (Nt=32)
__global__ void k_pack_wconv_frag(const float* __restrict__ w, unsigned short* __restrict__ dst) {
  int i = blockIdx.x * blockDim.x + threadIdx.x;
  if (i >= 2304 * 512) return;
  int k = i >> 9, n = i & 511;
  dst[b_frag_idx(k, n, 32)] = f2bf(w[(n * 256 + (k & 255)) * 9 + (k >> 8)]);
}

// w [O][K] row-major -> B-fragment layout with zero pad to Nld columns
__global__ void k_pack_wtrans_frag(const float* __restrict__ w, unsigned short* __restrict__ dst,
                                   int O, int K, int Nld) {
  int i = blockIdx.x * blockDim.x + threadIdx.x;
  if (i >= K * Nld) return;
  int k = i / Nld, n = i - k * Nld;
  float v = (n < O) ? w[n * K + k] : 0.f;
  dst[b_frag_idx(k, n, Nld >> 4)] = f2bf(v);
}

// zero A-fragment pad rows [m0, m1) for K columns
__global__ void k_zero_afrag(unsigned short* __restrict__ dst, int m0, int m1, int K) {
  int i = blockIdx.x * blockDim.x + threadIdx.x;
  int rows = m1 - m0;
  if (i >= rows * K) return;
  int m = m0 + i / K, k = i - (i / K) * K;
  dst[a_frag_idx(m, k, K >> 5)] = 0;
}

// -------------------------------------------- RPN 3x3 conv, implicit im2col
// A staged via LDS in fragment layout (b128 both ways); B read fragment-packed from L2.
__global__ __launch_bounds__(128) void k_conv3x3_wmma(
    const unsigned short* __restrict__ F,   // NHWC bf16
    int H, int W,
    const unsigned short* __restrict__ Bf,  // frag-packed, Kt=72, Nt=32
    const float* __restrict__ bias,
    float* __restrict__ sh, int M) {
  __shared__ __align__(32) unsigned short lAf[2048];   // 4 sub-tiles x 32 lanes x 16
  const int tid  = threadIdx.x;
  const int lane = tid & 31, wv = tid >> 5;
  const int half = lane >> 4, lm = lane & 15;
  const int tM = blockIdx.x * 64;
  const int tN = blockIdx.y * 64;
  const v8f zero = {0.f, 0.f, 0.f, 0.f, 0.f, 0.f, 0.f, 0.f};
  v8f acc[4] = {zero, zero, zero, zero};
  const unsigned short* Bp = Bf + ((size_t)(tN >> 4) * 32 + lane) * 16;

  for (int kt = 0; kt < 72; ++kt) {
    const int kk = kt * 32;
    // stage A: 256 groups of 8 consecutive channels -> one b128 global load,
    // one b128 LDS store directly into fragment slots.
    for (int g = tid; g < 256; g += 128) {
      int row = g >> 2, kc0 = (g & 3) * 8;
      int gm = tM + row;
      v8us v = {0, 0, 0, 0, 0, 0, 0, 0};
      if (gm < M) {
        int k = kk + kc0;
        int tap = k >> 8, c0 = k & 255;
        int y = gm / W + tap / 3 - 1;
        int x = gm % W + tap % 3 - 1;
        if ((unsigned)y < (unsigned)H && (unsigned)x < (unsigned)W)
          v = *(const v8us*)(F + (((size_t)y * W + x) << 8) + c0);
      }
      int hl = (kc0 >> 3) & 1;
      int dst = (row >> 4) * 512 + (16 * hl + (row & 15)) * 16 + ((kc0 & 16) >> 1);
      *(v8us*)(lAf + dst) = v;
    }
    __syncthreads();
    FragAB a;
    a.u = *(const v16us*)(lAf + wv * 512 + lane * 16);
#pragma unroll
    for (int t = 0; t < 4; ++t) {
      FragAB b;
      b.u = *(const v16us*)(Bp + t * 512);
      acc[t] = __builtin_amdgcn_wmma_f32_16x16x32_bf16(false, a.b, false, b.b,
                                                       (short)0, acc[t], false, false);
    }
    Bp += 32 * 512;   // next kt block (Nt=32)
    __syncthreads();
  }
  // epilogue: bias + relu -> row-major f32
#pragma unroll
  for (int t = 0; t < 4; ++t) {
    int col = tN + t * 16 + lm;
    float bb = bias[col];
#pragma unroll
    for (int r = 0; r < 8; ++r) {
      int row = tM + 16 * wv + r + 8 * half;
      if (row < M) sh[(size_t)row * 512 + col] = fmaxf(acc[t][r] + bb, 0.f);
    }
  }
}

// ------------------------------------- LDS-free fragment-layout bf16 GEMM
// Af: A-fragment layout (Mt=ceil16 rows padded/zeroed), Bf: B-fragment layout.
// C row-major f32 [row<M][Nld]; optional bias/relu; optional bf16out written in
// A-fragment layout for the NEXT gemm (Ktn = Nld/32), zero for pad rows.
__global__ __launch_bounds__(128) void k_gemm_wmma_frag(
    const unsigned short* __restrict__ Af, const unsigned short* __restrict__ Bf,
    float* __restrict__ C, int M, int Nld, int K,
    const float* __restrict__ bias, int nbias, int relu,
    unsigned short* __restrict__ bf16out, int Ktn) {
  const int lane = threadIdx.x & 31, wv = threadIdx.x >> 5;
  const int half = lane >> 4, lm = lane & 15;
  const int mt = blockIdx.x * 4 + wv;
  const int tN = blockIdx.y * 64;
  const int Kt = K >> 5, Nt = Nld >> 4;
  const v8f zero = {0.f, 0.f, 0.f, 0.f, 0.f, 0.f, 0.f, 0.f};
  v8f acc[4] = {zero, zero, zero, zero};

  const unsigned short* Ap = Af + ((size_t)mt * Kt * 32 + lane) * 16;
  const unsigned short* Bp = Bf + ((size_t)(tN >> 4) * 32 + lane) * 16;
  const size_t bStep = (size_t)Nt * 512;

  for (int kt = 0; kt < Kt; ++kt) {
    if (kt + 1 < Kt) {
      __builtin_prefetch(Ap + 512, 0, 1);
      __builtin_prefetch(Bp + bStep, 0, 1);
    }
    FragAB a;
    a.u = *(const v16us*)Ap;
    Ap += 512;
#pragma unroll
    for (int t = 0; t < 4; ++t) {
      FragAB b;
      b.u = *(const v16us*)(Bp + t * 512);
      acc[t] = __builtin_amdgcn_wmma_f32_16x16x32_bf16(false, a.b, false, b.b,
                                                       (short)0, acc[t], false, false);
    }
    Bp += bStep;
  }
#pragma unroll
  for (int t = 0; t < 4; ++t) {
    int col = tN + t * 16 + lm;
    float bb = (col < nbias) ? bias[col] : 0.f;
#pragma unroll
    for (int r = 0; r < 8; ++r) {
      int row = mt * 16 + r + 8 * half;
      float v = acc[t][r] + bb;
      if (relu) v = fmaxf(v, 0.f);
      if (row < M) C[(size_t)row * Nld + col] = v;
      if (bf16out) bf16out[a_frag_idx(row, col, Ktn)] = f2bf(row < M ? v : 0.f);
    }
  }
}

// ------------------------------------------------------- RPN 1x1 heads ----
__global__ void k_rpn_heads(const float* __restrict__ sh,
                            const float* __restrict__ wc, const float* __restrict__ bc,
                            const float* __restrict__ wb, const float* __restrict__ bb,
                            float* __restrict__ out_logits, float* __restrict__ out_bbox) {
  int idx = blockIdx.x * blockDim.x + threadIdx.x;
  if (idx >= 21824 * 18) return;
  int p = idx / 18, o = idx - p * 18;
  const float* w = (o < 6) ? (wc + o * 512) : (wb + (o - 6) * 512);
  const float* s = sh + (size_t)p * 512;
  float acc = 0.f;
  for (int c = 0; c < 512; ++c) acc += s[c] * w[c];
  if (o < 6) {
    int r = o >> 1, j = o & 1;
    out_logits[(3 * p + r) * 2 + j] = acc + bc[o];
  } else {
    int ob = o - 6, r = ob >> 2, j = ob & 3;
    out_bbox[(3 * p + r) * 4 + j] = acc + bb[ob];
  }
}

__global__ void k_scores(const float* __restrict__ logits, float* __restrict__ swork, int A) {
  int a = blockIdx.x * blockDim.x + threadIdx.x;
  if (a >= A) return;
  float l0 = logits[2 * a], l1 = logits[2 * a + 1];
  float m = fmaxf(l0, l1);
  float e0 = expf(l0 - m), e1 = expf(l1 - m);
  swork[a] = e1 / (e0 + e1);
}

// ------------------------------------------------- iterative exact top-k ----
__global__ __launch_bounds__(1024) void k_topk(float* __restrict__ work, int n, int k,
                                               int* __restrict__ oidx, float* __restrict__ oval) {
  __shared__ float sv[1024];
  __shared__ int   si[1024];
  int tid = threadIdx.x;
  for (int it = 0; it < k; ++it) {
    float best = -3.4e38f; int bi = n;
    for (int j = tid; j < n; j += 1024) {
      float v = work[j];
      if (v > best || (v == best && j < bi)) { best = v; bi = j; }
    }
    sv[tid] = best; si[tid] = bi;
    __syncthreads();
    for (int s = 512; s > 0; s >>= 1) {
      if (tid < s) {
        float v2 = sv[tid + s]; int i2 = si[tid + s];
        if (v2 > sv[tid] || (v2 == sv[tid] && i2 < si[tid])) { sv[tid] = v2; si[tid] = i2; }
      }
      __syncthreads();
    }
    if (tid == 0) { oidx[it] = si[0]; oval[it] = sv[0]; work[si[0]] = -3.4e38f; }
    __syncthreads();
  }
}

__global__ void k_gather_boxes(const int* __restrict__ tidx, const float* __restrict__ anchors,
                               const float* __restrict__ bbox, float* __restrict__ boxes) {
  int i = blockIdx.x * blockDim.x + threadIdx.x;
  if (i >= 2000) return;
  int a = tidx[i];
  float y1 = anchors[a * 4], x1 = anchors[a * 4 + 1], y2 = anchors[a * 4 + 2], x2 = anchors[a * 4 + 3];
  float d0 = bbox[a * 4] * 0.1f, d1 = bbox[a * 4 + 1] * 0.1f;
  float d2 = bbox[a * 4 + 2] * 0.2f, d3 = bbox[a * 4 + 3] * 0.2f;
  float h = y2 - y1, w = x2 - x1;
  float cy = y1 + 0.5f * h + d0 * h, cx = x1 + 0.5f * w + d1 * w;
  h *= expf(d2); w *= expf(d3);
  float b[4] = {cy - 0.5f * h, cx - 0.5f * w, cy + 0.5f * h, cx + 0.5f * w};
  for (int d = 0; d < 4; ++d) boxes[i * 4 + d] = fminf(fmaxf(b[d], 0.f), 1.f);
}

__global__ __launch_bounds__(1024) void k_nms(const float* __restrict__ b, int n, int* __restrict__ keep) {
  int tid = threadIdx.x;
  for (int j = tid; j < n; j += 1024) keep[j] = 1;
  __syncthreads();
  for (int i = 0; i < n; ++i) {
    if (keep[i]) {
      float y1i = b[i * 4], x1i = b[i * 4 + 1], y2i = b[i * 4 + 2], x2i = b[i * 4 + 3];
      float ai = (y2i - y1i) * (x2i - x1i);
      for (int j = tid; j < n; j += 1024) {
        if (j > i) {
          float yy1 = fmaxf(y1i, b[j * 4]),     xx1 = fmaxf(x1i, b[j * 4 + 1]);
          float yy2 = fminf(y2i, b[j * 4 + 2]), xx2 = fminf(x2i, b[j * 4 + 3]);
          float inter = fmaxf(yy2 - yy1, 0.f) * fmaxf(xx2 - xx1, 0.f);
          float aj = (b[j * 4 + 2] - b[j * 4]) * (b[j * 4 + 3] - b[j * 4 + 1]);
          float iou = inter / (ai + aj - inter + 1e-8f);
          if (iou > 0.7f) keep[j] = 0;
        }
      }
    }
    __syncthreads();
  }
}

__global__ void k_prep_post(const int* __restrict__ keep, const float* __restrict__ tval,
                            float* __restrict__ s2, int n) {
  int i = blockIdx.x * blockDim.x + threadIdx.x;
  if (i < n) s2[i] = keep[i] ? tval[i] : -1.0f;
}

__global__ void k_write_rois(const int* __restrict__ pick, const int* __restrict__ keep,
                             const float* __restrict__ boxes,
                             float* __restrict__ o_rois, float* __restrict__ roisws) {
  int i = blockIdx.x * blockDim.x + threadIdx.x;
  if (i >= 1000) return;
  int j = pick[i];
  float kf = keep[j] ? 1.f : 0.f;
  for (int d = 0; d < 4; ++d) {
    float v = boxes[j * 4 + d] * kf;
    o_rois[i * 4 + d] = v;
    roisws[i * 4 + d] = v;
  }
}

// ------------------- ROI align (crop-resize) -> bf16 A-fragment GEMM rows ----
__global__ __launch_bounds__(256) void k_roialign(
    const float* __restrict__ rois,
    const float* __restrict__ p2, const float* __restrict__ p3,
    const float* __restrict__ p4, const float* __restrict__ p5,
    unsigned short* __restrict__ A1) {
  int i = blockIdx.x;
  int c = threadIdx.x;
  float y1 = rois[i * 4], x1 = rois[i * 4 + 1], y2 = rois[i * 4 + 2], x2 = rois[i * 4 + 3];
  float h = y2 - y1, w = x2 - x1;
  float lv = 4.f + log2f(sqrtf(fmaxf(h * w, 1e-12f)) / 0.4375f);
  lv = fminf(fmaxf(rintf(lv), 2.f), 5.f);
  int li = (int)lv - 2;
  const float* fp = (li == 0) ? p2 : (li == 1) ? p3 : (li == 2) ? p4 : p5;
  int H = 128 >> li;
  const float* f = fp + (size_t)c * H * H;
  for (int py = 0; py < 7; ++py) {
    float ys = (y1 + h * (py * (1.f / 6.f))) * (float)(H - 1);
    float y0 = fminf(fmaxf(floorf(ys), 0.f), (float)(H - 1));
    float wy = ys - y0;
    int y0i = (int)y0;
    int y1i = (y0i + 1 < H - 1) ? (y0i + 1) : (H - 1);
    for (int px = 0; px < 7; ++px) {
      float xs = (x1 + w * (px * (1.f / 6.f))) * (float)(H - 1);
      float x0 = fminf(fmaxf(floorf(xs), 0.f), (float)(H - 1));
      float wx = xs - x0;
      int x0i = (int)x0;
      int x1i = (x0i + 1 < H - 1) ? (x0i + 1) : (H - 1);
      float v00 = f[y0i * H + x0i], v01 = f[y0i * H + x1i];
      float v10 = f[y1i * H + x0i], v11 = f[y1i * H + x1i];
      float val = v00 * (1.f - wy) * (1.f - wx) + v01 * (1.f - wy) * wx +
                  v10 * wy * (1.f - wx) + v11 * wy * wx;
      int k = c * 49 + py * 7 + px;
      A1[a_frag_idx(i, k, 392)] = f2bf(val);
    }
  }
}

// ------------------------------------------------------- final outputs ----
__global__ void k_cls_out(const float* __restrict__ L, float* __restrict__ cl, float* __restrict__ pr) {
  int i = blockIdx.x * blockDim.x + threadIdx.x;
  if (i >= 1000) return;
  const float* l = L + (size_t)i * 128;
  float mx = -3.4e38f;
  for (int j = 0; j < 81; ++j) { float v = l[j]; cl[i * 81 + j] = v; mx = fmaxf(mx, v); }
  float s = 0.f;
  for (int j = 0; j < 81; ++j) s += expf(l[j] - mx);
  float inv = 1.f / s;
  for (int j = 0; j < 81; ++j) pr[i * 81 + j] = expf(l[j] - mx) * inv;
}

__global__ void k_dl_out(const float* __restrict__ D, float* __restrict__ dl) {
  int idx = blockIdx.x * blockDim.x + threadIdx.x;
  if (idx >= 1000 * 324) return;
  int i = idx / 324, j = idx - i * 324;
  dl[idx] = D[(size_t)i * 384 + j];
}

// ---------------------------------------------------------------- host ----
extern "C" void kernel_launch(void* const* d_in, const int* in_sizes, int n_in,
                              void* d_out, int out_size, void* d_ws, size_t ws_size,
                              hipStream_t stream) {
  (void)in_sizes; (void)n_in; (void)out_size; (void)ws_size;
  const float* p[5];
  for (int i = 0; i < 5; ++i) p[i] = (const float*)d_in[i];
  const float* anchors    = (const float*)d_in[5];
  const float* rpn_conv_w = (const float*)d_in[6];
  const float* rpn_cls_w  = (const float*)d_in[7];
  const float* rpn_bbox_w = (const float*)d_in[8];
  const float* cls_w1     = (const float*)d_in[9];
  const float* cls_w2     = (const float*)d_in[10];
  const float* cls_wl     = (const float*)d_in[11];
  const float* cls_wd     = (const float*)d_in[12];
  const float* rpn_conv_b = (const float*)d_in[13];
  const float* rpn_cls_b  = (const float*)d_in[14];
  const float* rpn_bbox_b = (const float*)d_in[15];
  const float* cls_b1     = (const float*)d_in[16];
  const float* cls_b2     = (const float*)d_in[17];
  const float* cls_bl     = (const float*)d_in[18];
  const float* cls_bd     = (const float*)d_in[19];

  float* out      = (float*)d_out;
  float* o_logits = out;            // 65472*2
  float* o_bbox   = out + 130944;   // 65472*4
  float* o_rois   = out + 392832;   // 1000*4
  float* o_cl     = out + 396832;   // 1000*81
  float* o_pr     = out + 477832;   // 1000*81
  float* o_dl     = out + 558832;   // 1000*324

  char* wsb = (char*)d_ws;
  size_t off = 0;
  auto alloc = [&](size_t bytes) -> void* {
    void* q = wsb + off;
    off = (off + bytes + 255) & ~(size_t)255;
    return q;
  };

  static const int HS[5] = {128, 64, 32, 16, 8};
  int Ms[5], featoff[5], shbase[5];
  {
    int fo = 0, sb = 0;
    for (int l = 0; l < 5; ++l) { Ms[l] = HS[l] * HS[l]; featoff[l] = fo; fo += 256 * Ms[l]; shbase[l] = sb; sb += Ms[l]; }
  }

  unsigned short* featbf = (unsigned short*)alloc(sizeof(unsigned short) * (size_t)256 * 21824);
  unsigned short* BwC    = (unsigned short*)alloc(sizeof(unsigned short) * (size_t)2304 * 512);
  float*          sh     = (float*)alloc(sizeof(float) * (size_t)21824 * 512);
  float*          swork  = (float*)alloc(sizeof(float) * 65472);
  int*            tidx   = (int*)alloc(sizeof(int) * 2000);
  float*          tval   = (float*)alloc(sizeof(float) * 2000);
  float*          boxes  = (float*)alloc(sizeof(float) * 2000 * 4);
  int*            keep   = (int*)alloc(sizeof(int) * 2000);
  float*          s2     = (float*)alloc(sizeof(float) * 2000);
  int*            pick   = (int*)alloc(sizeof(int) * 1000);
  float*          pickv  = (float*)alloc(sizeof(float) * 1000);
  float*          roisws = (float*)alloc(sizeof(float) * 1000 * 4);
  unsigned short* A1     = (unsigned short*)alloc(sizeof(unsigned short) * (size_t)1024 * 12544);
  unsigned short* Bw1    = (unsigned short*)alloc(sizeof(unsigned short) * (size_t)12544 * 1024);
  float*          X1f    = (float*)alloc(sizeof(float) * (size_t)1000 * 1024);
  unsigned short* X1b    = (unsigned short*)alloc(sizeof(unsigned short) * (size_t)1024 * 1024);
  unsigned short* Bw2    = (unsigned short*)alloc(sizeof(unsigned short) * (size_t)1024 * 1024);
  float*          X2f    = (float*)alloc(sizeof(float) * (size_t)1000 * 1024);
  unsigned short* X2b    = (unsigned short*)alloc(sizeof(unsigned short) * (size_t)1024 * 1024);
  unsigned short* Bwl    = (unsigned short*)alloc(sizeof(unsigned short) * (size_t)1024 * 128);
  unsigned short* Bwd    = (unsigned short*)alloc(sizeof(unsigned short) * (size_t)1024 * 384);
  float*          L128   = (float*)alloc(sizeof(float) * (size_t)1000 * 128);
  float*          D384   = (float*)alloc(sizeof(float) * (size_t)1000 * 384);

  // 1) pack features (NHWC bf16) + all weights (fragment layout)
  for (int l = 0; l < 5; ++l) {
    int n = 256 * Ms[l];
    k_pack_nhwc<<<(n + 255) / 256, 256, 0, stream>>>(p[l], featbf + featoff[l], Ms[l]);
  }
  k_pack_wconv_frag<<<(2304 * 512 + 255) / 256, 256, 0, stream>>>(rpn_conv_w, BwC);
  k_pack_wtrans_frag<<<(12544 * 1024 + 255) / 256, 256, 0, stream>>>(cls_w1, Bw1, 1024, 12544, 1024);
  k_pack_wtrans_frag<<<(1024 * 1024 + 255) / 256, 256, 0, stream>>>(cls_w2, Bw2, 1024, 1024, 1024);
  k_pack_wtrans_frag<<<(1024 * 128 + 255) / 256, 256, 0, stream>>>(cls_wl, Bwl, 81, 1024, 128);
  k_pack_wtrans_frag<<<(1024 * 384 + 255) / 256, 256, 0, stream>>>(cls_wd, Bwd, 324, 1024, 384);

  // 2) RPN 3x3 conv via WMMA (fused bias+relu)
  for (int l = 0; l < 5; ++l) {
    dim3 g((Ms[l] + 63) / 64, 8);
    k_conv3x3_wmma<<<g, 128, 0, stream>>>(featbf + featoff[l], HS[l], HS[l], BwC,
                                          rpn_conv_b, sh + (size_t)shbase[l] * 512, Ms[l]);
  }

  // 3) RPN 1x1 heads -> rpn_logits / rpn_bbox outputs, then objectness scores
  k_rpn_heads<<<(21824 * 18 + 255) / 256, 256, 0, stream>>>(sh, rpn_cls_w, rpn_cls_b,
                                                            rpn_bbox_w, rpn_bbox_b, o_logits, o_bbox);
  k_scores<<<(65472 + 255) / 256, 256, 0, stream>>>(o_logits, swork, 65472);

  // 4) proposals: top-2000 -> decode -> NMS -> top-1000
  k_topk<<<1, 1024, 0, stream>>>(swork, 65472, 2000, tidx, tval);
  k_gather_boxes<<<(2000 + 255) / 256, 256, 0, stream>>>(tidx, anchors, o_bbox, boxes);
  k_nms<<<1, 1024, 0, stream>>>(boxes, 2000, keep);
  k_prep_post<<<(2000 + 255) / 256, 256, 0, stream>>>(keep, tval, s2, 2000);
  k_topk<<<1, 1024, 0, stream>>>(s2, 2000, 1000, pick, pickv);
  k_write_rois<<<(1000 + 255) / 256, 256, 0, stream>>>(pick, keep, boxes, o_rois, roisws);

  // 5) ROI align straight into A-fragment layout; zero pad rows 1000..1023
  k_roialign<<<1000, 256, 0, stream>>>(roisws, p[0], p[1], p[2], p[3], A1);
  k_zero_afrag<<<(24 * 12544 + 255) / 256, 256, 0, stream>>>(A1, 1000, 1024, 12544);

  // 6) classifier head: four LDS-free fragment GEMMs
  {
    dim3 g1(16, 16); // 1024 rows x 1024 cols
    k_gemm_wmma_frag<<<g1, 128, 0, stream>>>(A1, Bw1, X1f, 1000, 1024, 12544,
                                             cls_b1, 1024, 1, X1b, 32);
    k_gemm_wmma_frag<<<g1, 128, 0, stream>>>(X1b, Bw2, X2f, 1000, 1024, 1024,
                                             cls_b2, 1024, 1, X2b, 32);
  }
  {
    dim3 gl(16, 2);  // 128 padded cols
    k_gemm_wmma_frag<<<gl, 128, 0, stream>>>(X2b, Bwl, L128, 1000, 128, 1024,
                                             cls_bl, 81, 0, (unsigned short*)nullptr, 0);
    dim3 gd(16, 6);  // 384 padded cols
    k_gemm_wmma_frag<<<gd, 128, 0, stream>>>(X2b, Bwd, D384, 1000, 384, 1024,
                                             cls_bd, 324, 0, (unsigned short*)nullptr, 0);
  }

  // 7) logits copy + softmax, deltas copy
  k_cls_out<<<(1000 + 255) / 256, 256, 0, stream>>>(L128, o_cl, o_pr);
  k_dl_out<<<(1000 * 324 + 255) / 256, 256, 0, stream>>>(D384, o_dl);
}
